// SparseConv2d_65309272703017
// MI455X (gfx1250) — matrix operand order repaired and checked by reading, not compiled, not run on hardware
//
#include <hip/hip_runtime.h>

typedef __bf16 bf16_t;
typedef __attribute__((ext_vector_type(16))) __bf16 v16bf;
typedef __attribute__((ext_vector_type(8)))  __bf16 v8bf;
typedef __attribute__((ext_vector_type(8)))  float  v8f;

#define C_IN    64
#define HW_DIM  128
#define K_OUT   128
#define XSTRIDE 72   // padded channel-stride for LDS bank spread (72*2B = 144B)

// ---------------------------------------------------------------------------
// Kernel 1: repack weights (128,64,3,3) OIHW fp32 -> bf16 hi/lo, layout
// [rs][m][c] so A-fragments are contiguous along c (the GEMM K dim).
// ---------------------------------------------------------------------------
__global__ __launch_bounds__(256) void repack_w_kernel(
    const float* __restrict__ w, bf16_t* __restrict__ whi, bf16_t* __restrict__ wlo) {
  int i = blockIdx.x * 256 + threadIdx.x;
  if (i >= 9 * K_OUT * C_IN) return;
  int c  = i & 63;
  int m  = (i >> 6) & 127;
  int rs = i >> 13;
  int r = rs / 3, s = rs % 3;
  float v = w[(m * C_IN + c) * 9 + r * 3 + s];
  bf16_t hi = (bf16_t)v;
  bf16_t lo = (bf16_t)(v - (float)hi);
  whi[i] = hi;
  wlo[i] = lo;
}

// ---------------------------------------------------------------------------
// Kernel 2: shift-decomposed implicit GEMM conv, bf16x3 WMMA, f32 accumulate.
// Block = 256 threads (8 waves). Block tile: all 128 out-channels x 64 pixels
// of one output row. Wave tile: 32 channels x 32 pixels (2x2 WMMA tiles).
// ---------------------------------------------------------------------------
__global__ __launch_bounds__(256) void conv3x3_wmma_kernel(
    const float*  __restrict__ x,
    const bf16_t* __restrict__ whi,
    const bf16_t* __restrict__ wlo,
    const float*  __restrict__ bias,
    float*        __restrict__ out) {

  __shared__ bf16_t XH[3 * 66 * XSTRIDE];
  __shared__ bf16_t XL[3 * 66 * XSTRIDE];

  const int tid = threadIdx.x;
  const int bx  = blockIdx.x;
  const int w0  = (bx & 1) * 64;        // pixel-tile start in w
  const int h   = (bx >> 1) & 127;      // output row
  const int n   = bx >> 8;              // image

  // ---- Stage x halo slab: rows h-1..h+1, cols w0-1..w0+64, 64 ch, bf16 hi/lo,
  //      K(channel)-major in LDS so B-fragment loads are b128s. Coalesced on w.
  for (int i = tid; i < 3 * 66 * C_IN; i += 256) {
    int wi = i % 66;
    int c  = (i / 66) & 63;
    int r  = i / (66 * C_IN);
    int hin = h + r - 1;
    int win = w0 + wi - 1;
    float v = 0.0f;
    if ((unsigned)hin < 128u && (unsigned)win < 128u)
      v = x[((n * C_IN + c) * HW_DIM + hin) * HW_DIM + win];
    bf16_t hi = (bf16_t)v;
    bf16_t lo = (bf16_t)(v - (float)hi);
    int di = (r * 66 + wi) * XSTRIDE + c;
    XH[di] = hi;
    XL[di] = lo;
  }
  __syncthreads();

  const int lane   = tid & 31;
  const int waveId = tid >> 5;
  const int m0 = (waveId & 3) * 32;     // channel tile base (4 waves across M)
  const int p0 = (waveId >> 2) * 32;    // pixel tile base   (2 waves across N)

  const int hl    = lane >> 4;          // 0: lanes 0-15, 1: lanes 16-31
  const int l15   = lane & 15;
  const int koffA = hl * 8;             // A: K pairs {0..7,16..23} / {8..15,24..31}
  const int koffB = hl * 8;             // B: K pairs {0..7,16..23} / {8..15,24..31}

  v8f acc[2][2] = {};

  for (int r = 0; r < 3; ++r) {
#pragma unroll
    for (int s = 0; s < 3; ++s) {
      const int rs = r * 3 + s;
#pragma unroll
      for (int kc = 0; kc < 2; ++kc) {
        const int c0 = kc * 32;

        // ---- A fragments (weights) straight from L2-resident repacked buffer
        v16bf ahi[2], alo[2];
#pragma unroll
        for (int mt = 0; mt < 2; ++mt) {
          int m = m0 + mt * 16 + l15;
          int bi = (rs * K_OUT + m) * C_IN + c0 + koffA;
          v8bf h0 = *(const v8bf*)(whi + bi);
          v8bf h1 = *(const v8bf*)(whi + bi + 16);
          ahi[mt] = __builtin_shufflevector(h0, h1, 0,1,2,3,4,5,6,7,8,9,10,11,12,13,14,15);
          v8bf l0 = *(const v8bf*)(wlo + bi);
          v8bf l1 = *(const v8bf*)(wlo + bi + 16);
          alo[mt] = __builtin_shufflevector(l0, l1, 0,1,2,3,4,5,6,7,8,9,10,11,12,13,14,15);
        }

        // ---- B fragments (activations) from LDS, shifted by s in w
        v16bf bhi[2], blo[2];
#pragma unroll
        for (int nt = 0; nt < 2; ++nt) {
          int p  = p0 + nt * 16 + l15;          // output pixel within block tile
          int wi = p + s;                        // halo index (0 == w0-1)
          int di = (r * 66 + wi) * XSTRIDE + c0 + koffB;
          v8bf h0 = *(const v8bf*)(XH + di);
          v8bf h1 = *(const v8bf*)(XH + di + 16);
          bhi[nt] = __builtin_shufflevector(h0, h1, 0,1,2,3,4,5,6,7,8,9,10,11,12,13,14,15);
          v8bf l0 = *(const v8bf*)(XL + di);
          v8bf l1 = *(const v8bf*)(XL + di + 16);
          blo[nt] = __builtin_shufflevector(l0, l1, 0,1,2,3,4,5,6,7,8,9,10,11,12,13,14,15);
        }

        // ---- bf16x3: hi*hi + hi*lo + lo*hi (f32 accumulate)
#pragma unroll
        for (int mt = 0; mt < 2; ++mt) {
#pragma unroll
          for (int nt = 0; nt < 2; ++nt) {
            acc[mt][nt] = __builtin_amdgcn_wmma_f32_16x16x32_bf16(
                false, ahi[mt], false, bhi[nt], (short)0, acc[mt][nt], false, false);
            acc[mt][nt] = __builtin_amdgcn_wmma_f32_16x16x32_bf16(
                false, ahi[mt], false, blo[nt], (short)0, acc[mt][nt], false, false);
            acc[mt][nt] = __builtin_amdgcn_wmma_f32_16x16x32_bf16(
                false, alo[mt], false, bhi[nt], (short)0, acc[mt][nt], false, false);
          }
        }
      }
    }
  }

  // ---- Epilogue: bias add + store. C/D layout: VGPR v -> M = v + 8*hl, N = l15.
#pragma unroll
  for (int mt = 0; mt < 2; ++mt) {
#pragma unroll
    for (int nt = 0; nt < 2; ++nt) {
      int px   = p0 + nt * 16 + l15;
      int wout = w0 + px;
#pragma unroll
      for (int vg = 0; vg < 8; ++vg) {
        int M = m0 + mt * 16 + vg + hl * 8;
        out[((n * K_OUT + M) * HW_DIM + h) * HW_DIM + wout] = acc[mt][nt][vg] + bias[M];
      }
    }
  }
}

// ---------------------------------------------------------------------------
extern "C" void kernel_launch(void* const* d_in, const int* in_sizes, int n_in,
                              void* d_out, int out_size, void* d_ws, size_t ws_size,
                              hipStream_t stream) {
  const float* x    = (const float*)d_in[0];  // (32,64,128,128)
  const float* w    = (const float*)d_in[1];  // (128,64,3,3)
  const float* bias = (const float*)d_in[2];  // (128,)
  float* out = (float*)d_out;

  bf16_t* whi = (bf16_t*)d_ws;                // 9*128*64 bf16
  bf16_t* wlo = whi + 9 * K_OUT * C_IN;       // 9*128*64 bf16 (total 288 KB ws)

  const int wElems = 9 * K_OUT * C_IN;        // 73728
  repack_w_kernel<<<(wElems + 255) / 256, 256, 0, stream>>>(w, whi, wlo);

  // 32 images * 128 rows * 2 pixel-tiles = 8192 blocks
  conv3x3_wmma_kernel<<<8192, 256, 0, stream>>>(x, whi, wlo, bias, out);
}
